// LongFormer_23862838297260
// MI455X (gfx1250) — matrix operand order repaired and checked by reading, not compiled, not run on hardware
//
#include <hip/hip_runtime.h>
#include <hip/hip_bf16.h>
#include <math.h>

// Problem constants
#define B_    2
#define S_    1024
#define HID_  768
#define NH_   12
#define DH_   64
#define FF_   3072
#define L_    12
#define W_    256
#define NCLS_ 15
#define M_    (B_ * S_)        // 2048 rows
#define SP_   (S_ + 2 * W_)    // 1536 padded key positions
#define NB_   (S_ / W_)        // 4 query blocks

typedef __attribute__((ext_vector_type(16))) __bf16        v16bf;
typedef __attribute__((ext_vector_type(8)))  float         v8f;
typedef __attribute__((ext_vector_type(4)))  unsigned int  u32x4;

union FragBF {
  v16bf v;
  u32x4 q[2];
  unsigned short us[16];
};

__device__ __forceinline__ unsigned short f2bf(float f) {
  union { float f; unsigned int u; } x; x.f = f;
  unsigned int u = x.u;
  u += 0x7fffu + ((u >> 16) & 1u);   // round-to-nearest-even
  return (unsigned short)(u >> 16);
}

// ---------------------------------------------------------------------------
// Dense GEMM: C[M,N] = A[M,K](bf16) * BT[N,K](bf16)^T + bias, fused epilogue.
// Wave32: each wave computes a 16(M) x 64(N) strip; 8 waves/WG stack along M.
// EPI: 0 = (acc+bias)*scale -> bf16 ; 1 = acc+bias -> f32 ;
//      2 = gelu(acc+bias) -> bf16   ; 3 = relu(acc+bias) -> f32
// ---------------------------------------------------------------------------
template <int EPI>
__global__ void __launch_bounds__(256)
wmma_gemm(const unsigned short* __restrict__ A,
          const unsigned short* __restrict__ BT,
          const float* __restrict__ bias,
          void* __restrict__ Cout,
          int M, int N, int K, float scale)
{
  const int lane = threadIdx.x & 31;
  // wave index is wave-uniform: keep it in an SGPR
  const int wave = __builtin_amdgcn_readfirstlane(threadIdx.x >> 5);
  const int half = lane >> 4;
  const int l16  = lane & 15;
  const int m0 = blockIdx.y * 128 + wave * 16;
  const int n0 = blockIdx.x * 64;

  v8f acc[4] = {};

  const unsigned short* arow = A + (size_t)(m0 + l16) * K;
#pragma unroll 2
  for (int k0 = 0; k0 < K; k0 += 32) {
    FragBF af;
    af.q[0] = *(const u32x4*)(arow + k0 + half * 8);
    af.q[1] = *(const u32x4*)(arow + k0 + half * 8 + 16);
#pragma unroll
    for (int t = 0; t < 4; ++t) {
      const unsigned short* bcol =
          BT + (size_t)(n0 + t * 16 + l16) * K + k0 + half * 8;
      FragBF bf;
      bf.q[0] = *(const u32x4*)(bcol);
      bf.q[1] = *(const u32x4*)(bcol + 16);
      acc[t] = __builtin_amdgcn_wmma_f32_16x16x32_bf16(
          false, af.v, false, bf.v, (short)0, acc[t], false, false);
    }
  }

#pragma unroll
  for (int t = 0; t < 4; ++t) {
    int col = n0 + t * 16 + l16;
    float bv = bias ? bias[col] : 0.f;
#pragma unroll
    for (int v = 0; v < 8; ++v) {
      int row = m0 + v + 8 * half;
      float x = acc[t][v] + bv;
      size_t o = (size_t)row * N + col;
      if (EPI == 0) {
        ((unsigned short*)Cout)[o] = f2bf(x * scale);
      } else if (EPI == 1) {
        ((float*)Cout)[o] = x;
      } else if (EPI == 2) {
        float g = 0.5f * x * (1.f + erff(x * 0.70710678118654752f));
        ((unsigned short*)Cout)[o] = f2bf(g);
      } else {
        ((float*)Cout)[o] = fmaxf(x, 0.f);
      }
    }
  }
}

// ---------------------------------------------------------------------------
// Sliding-window attention, flash-style online softmax.
// One wave handles one 16-query tile of one (b, head, block-of-256-queries).
// Q [M,HID] bf16 pre-scaled by 1/sqrt(DH); KP [B*NH, SP, DH] zero-padded;
// VT [B*NH, DH, SP] zero-padded transposed V; O [M,HID] bf16.
// All per-wave tile coordinates are forced scalar (readfirstlane) so the
// chunk loop is pure SALU control flow and EXEC stays all-ones for WMMA.
// All per-element masking is branchless (bitwise predicates + cndmask).
// ---------------------------------------------------------------------------
__global__ void __launch_bounds__(256)
sw_attn(const unsigned short* __restrict__ Q,
        const unsigned short* __restrict__ KP,
        const unsigned short* __restrict__ VT,
        const int* __restrict__ am,
        unsigned short* __restrict__ O)
{
  __shared__ float ss[8][16 * 33];
  const int lane = threadIdx.x & 31;
  const int wave = __builtin_amdgcn_readfirstlane(threadIdx.x >> 5);
  const int half = lane >> 4;
  const int l16  = lane & 15;

  const int wid = __builtin_amdgcn_readfirstlane(blockIdx.x * 8 + wave);
  const int qt  = wid & 15;
  const int c   = (wid >> 4) & 3;
  const int h   = (wid >> 6) % NH_;
  const int b   = wid / (16 * NB_ * NH_);
  const int i0  = qt * 16;

  // Q fragments for d = 0..31 and 32..63 (A layout)
  FragBF qf[2];
  {
    int sq = c * W_ + i0 + l16;
    const unsigned short* qrow = Q + (size_t)(b * S_ + sq) * HID_ + h * DH_;
#pragma unroll
    for (int ds = 0; ds < 2; ++ds) {
      qf[ds].q[0] = *(const u32x4*)(qrow + ds * 32 + half * 8);
      qf[ds].q[1] = *(const u32x4*)(qrow + ds * 32 + half * 8 + 16);
    }
  }
  const unsigned short* kpb = KP + (size_t)(b * NH_ + h) * SP_ * DH_;
  const unsigned short* vtb = VT + (size_t)(b * NH_ + h) * DH_ * SP_;
  const int* amb = am + b * S_;

  v8f accO[4] = {};
  float mrow = -1e30f, lrow = 0.f;
  float* myss = &ss[wave][0];

  // band for rows i0..i0+15 covers j in [i0, i0+15+2W]; chunk granularity 32
  const int c_lo = __builtin_amdgcn_readfirstlane(i0 >> 5);
  const int c_hi = __builtin_amdgcn_readfirstlane((i0 + 15 + 2 * W_) >> 5);

  for (int chunk = c_lo; chunk <= c_hi; ++chunk) {   // scalar loop bounds
    int jb = chunk * 32;

    // ----- scores: two 16x16 tiles (32 keys), contraction over d=64 -----
#pragma unroll
    for (int t = 0; t < 2; ++t) {
      v8f ct = {};
      int j = jb + t * 16 + l16;
      int kprow = c * W_ + j;
#pragma unroll
      for (int ds = 0; ds < 2; ++ds) {
        const unsigned short* kr = kpb + (size_t)kprow * DH_ + ds * 32 + half * 8;
        FragBF kf;
        kf.q[0] = *(const u32x4*)(kr);
        kf.q[1] = *(const u32x4*)(kr + 16);
        ct = __builtin_amdgcn_wmma_f32_16x16x32_bf16(
            false, qf[ds].v, false, kf.v, (short)0, ct, false, false);
      }
      // branchless validity: clamp the mask index, load unconditionally,
      // combine predicates with bitwise ops (no short-circuit -> no execz)
      int sk  = c * W_ + j - W_;
      int skc = min(max(sk, 0), S_ - 1);
      int mv  = amb[skc];
      int kvalid = (int)(sk >= 0) & (int)(sk < S_) & (int)(mv != 0);
#pragma unroll
      for (int v = 0; v < 8; ++v) {
        int i = i0 + v + 8 * half;
        int ok = kvalid & (int)(j >= i) & (int)(j <= i + 2 * W_);
        myss[(v + 8 * half) * 33 + t * 16 + l16] = ok ? ct[v] : -1e9f;
      }
    }
    asm volatile("s_wait_dscnt 0x0" ::: "memory");  // same-wave LDS RAW

    // ----- reload in A layout, online softmax -----
    float pv[16];
#pragma unroll
    for (int q2 = 0; q2 < 8; ++q2) {
      pv[q2]     = myss[l16 * 33 + half * 8 + q2];
      pv[q2 + 8] = myss[l16 * 33 + 16 + half * 8 + q2];
    }
    float rmax = -1e30f;
#pragma unroll
    for (int q2 = 0; q2 < 16; ++q2) rmax = fmaxf(rmax, pv[q2]);
    rmax = fmaxf(rmax, __shfl_xor(rmax, 16, 32));
    float newm  = fmaxf(mrow, rmax);
    float alpha = __expf(mrow - newm);
    float rs = 0.f;
#pragma unroll
    for (int q2 = 0; q2 < 16; ++q2) { pv[q2] = __expf(pv[q2] - newm); rs += pv[q2]; }
    rs += __shfl_xor(rs, 16, 32);
    lrow = lrow * alpha + rs;
    mrow = newm;

    // rescale accumulator (C layout: row = v + 8*half, alpha lives in lane row)
#pragma unroll
    for (int v = 0; v < 8; ++v) {
      float fa = __shfl(alpha, v + 8 * half, 32);
#pragma unroll
      for (int t = 0; t < 4; ++t) accO[t][v] *= fa;
    }

    // P fragment already in A layout
    FragBF pf;
#pragma unroll
    for (int q2 = 0; q2 < 16; ++q2) pf.us[q2] = f2bf(pv[q2]);

    // ----- PV: accumulate over the 32 keys of this chunk -----
#pragma unroll
    for (int n = 0; n < 4; ++n) {
      int d = n * 16 + l16;
      const unsigned short* vr =
          vtb + (size_t)d * SP_ + c * W_ + jb + half * 8;
      FragBF vf;
      vf.q[0] = *(const u32x4*)(vr);
      vf.q[1] = *(const u32x4*)(vr + 16);
      accO[n] = __builtin_amdgcn_wmma_f32_16x16x32_bf16(
          false, pf.v, false, vf.v, (short)0, accO[n], false, false);
    }
  }

  float linv = 1.f / lrow;
#pragma unroll
  for (int v = 0; v < 8; ++v) {
    float li = __shfl(linv, v + 8 * half, 32);
    int sq = c * W_ + i0 + v + 8 * half;
#pragma unroll
    for (int n = 0; n < 4; ++n) {
      O[(size_t)(b * S_ + sq) * HID_ + h * DH_ + n * 16 + l16] =
          f2bf(accO[n][v] * li);
    }
  }
}

// ---------------------------------------------------------------------------
// Embedding lookup + LayerNorm -> hf32, hbf16. One block per row.
// ---------------------------------------------------------------------------
__global__ void __launch_bounds__(256)
embed_ln(const int* __restrict__ ids, const float* __restrict__ we,
         const float* __restrict__ pe, const float* __restrict__ g,
         const float* __restrict__ be, float* __restrict__ hf,
         unsigned short* __restrict__ hb)
{
  __shared__ float red[256];
  int m = blockIdx.x, s = m % S_, tid = threadIdx.x;
  int id = ids[m];
  float x[3];
#pragma unroll
  for (int r = 0; r < 3; ++r) {
    int j = tid + r * 256;
    x[r] = we[(size_t)id * HID_ + j] + pe[(size_t)s * HID_ + j];
  }
  float sum = x[0] + x[1] + x[2];
  red[tid] = sum; __syncthreads();
  for (int o = 128; o > 0; o >>= 1) { if (tid < o) red[tid] += red[tid + o]; __syncthreads(); }
  float mean = red[0] * (1.f / HID_); __syncthreads();
  float s2 = 0.f;
#pragma unroll
  for (int r = 0; r < 3; ++r) { float d = x[r] - mean; s2 += d * d; }
  red[tid] = s2; __syncthreads();
  for (int o = 128; o > 0; o >>= 1) { if (tid < o) red[tid] += red[tid + o]; __syncthreads(); }
  float rstd = rsqrtf(red[0] * (1.f / HID_) + 1e-5f);
#pragma unroll
  for (int r = 0; r < 3; ++r) {
    int j = tid + r * 256;
    float y = (x[r] - mean) * rstd * g[j] + be[j];
    hf[(size_t)m * HID_ + j] = y;
    hb[(size_t)m * HID_ + j] = f2bf(y);
  }
}

// Residual add + LayerNorm (in place on hf), also refresh bf16 copy.
__global__ void __launch_bounds__(256)
add_ln(float* __restrict__ hf, const float* __restrict__ pj,
       const float* __restrict__ g, const float* __restrict__ be,
       unsigned short* __restrict__ hb)
{
  __shared__ float red[256];
  int m = blockIdx.x, tid = threadIdx.x;
  float x[3];
#pragma unroll
  for (int r = 0; r < 3; ++r) {
    int j = tid + r * 256;
    x[r] = hf[(size_t)m * HID_ + j] + pj[(size_t)m * HID_ + j];
  }
  float sum = x[0] + x[1] + x[2];
  red[tid] = sum; __syncthreads();
  for (int o = 128; o > 0; o >>= 1) { if (tid < o) red[tid] += red[tid + o]; __syncthreads(); }
  float mean = red[0] * (1.f / HID_); __syncthreads();
  float s2 = 0.f;
#pragma unroll
  for (int r = 0; r < 3; ++r) { float d = x[r] - mean; s2 += d * d; }
  red[tid] = s2; __syncthreads();
  for (int o = 128; o > 0; o >>= 1) { if (tid < o) red[tid] += red[tid + o]; __syncthreads(); }
  float rstd = rsqrtf(red[0] * (1.f / HID_) + 1e-5f);
#pragma unroll
  for (int r = 0; r < 3; ++r) {
    int j = tid + r * 256;
    float y = (x[r] - mean) * rstd * g[j] + be[j];
    hf[(size_t)m * HID_ + j] = y;
    hb[(size_t)m * HID_ + j] = f2bf(y);
  }
}

// W[K,N] f32 -> WT[N,K] bf16
__global__ void transpose_conv(const float* __restrict__ w,
                               unsigned short* __restrict__ wt, int K, int N)
{
  unsigned int i = blockIdx.x * blockDim.x + threadIdx.x;
  if (i >= (unsigned int)(K * N)) return;
  int n = i / K, k = i % K;
  wt[i] = f2bf(w[(size_t)k * N + n]);
}

// k bf16 [M,HID] -> KP [B*NH, SP, DH] zero-padded
__global__ void pad_k(const unsigned short* __restrict__ kb,
                      unsigned short* __restrict__ kp)
{
  unsigned int i = blockIdx.x * blockDim.x + threadIdx.x;
  if (i >= (unsigned int)(B_ * NH_ * SP_ * DH_)) return;
  int d  = i % DH_;
  int sp = (i / DH_) % SP_;
  int bh = i / (DH_ * SP_);
  int b = bh / NH_, h = bh % NH_;
  int s = sp - W_;
  unsigned short v = 0;
  if (s >= 0 && s < S_) v = kb[(size_t)(b * S_ + s) * HID_ + h * DH_ + d];
  kp[i] = v;
}

// v bf16 [M,HID] -> VT [B*NH, DH, SP] zero-padded transposed
__global__ void transpose_vp(const unsigned short* __restrict__ vb,
                             unsigned short* __restrict__ vt)
{
  unsigned int i = blockIdx.x * blockDim.x + threadIdx.x;
  if (i >= (unsigned int)(B_ * NH_ * DH_ * SP_)) return;
  int sp = i % SP_;
  int d  = (i / SP_) % DH_;
  int bh = i / (SP_ * DH_);
  int b = bh / NH_, h = bh % NH_;
  int s = sp - W_;
  unsigned short v = 0;
  if (s >= 0 && s < S_) v = vb[(size_t)(b * S_ + s) * HID_ + h * DH_ + d];
  vt[i] = v;
}

// r1 [M,256] f32 @ hW2 [256,15] + hb2 -> out [M,15] f32
__global__ void __launch_bounds__(64)
head2(const float* __restrict__ r1, const float* __restrict__ w2,
      const float* __restrict__ b2, float* __restrict__ out)
{
  __shared__ float row[256];
  int m = blockIdx.x, tid = threadIdx.x;
#pragma unroll
  for (int r = 0; r < 4; ++r) row[tid + r * 64] = r1[(size_t)m * 256 + tid + r * 64];
  __syncthreads();
  if (tid < NCLS_) {
    float acc = b2[tid];
    for (int j = 0; j < 256; ++j) acc += row[j] * w2[j * NCLS_ + tid];
    out[(size_t)m * NCLS_ + tid] = acc;
  }
}

// ---------------------------------------------------------------------------
extern "C" void kernel_launch(void* const* d_in, const int* in_sizes, int n_in,
                              void* d_out, int out_size, void* d_ws, size_t ws_size,
                              hipStream_t stream)
{
  (void)in_sizes; (void)n_in; (void)out_size; (void)ws_size;
  const int*   ids  = (const int*)d_in[0];
  const int*   am   = (const int*)d_in[1];
  const float* we   = (const float*)d_in[2];
  const float* pe   = (const float*)d_in[3];
  const float* eg   = (const float*)d_in[4];
  const float* eb   = (const float*)d_in[5];
  const float* Wq   = (const float*)d_in[6];
  const float* bq   = (const float*)d_in[7];
  const float* Wk   = (const float*)d_in[8];
  const float* bk   = (const float*)d_in[9];
  const float* Wv   = (const float*)d_in[10];
  const float* bv   = (const float*)d_in[11];
  const float* Wo   = (const float*)d_in[12];
  const float* bo   = (const float*)d_in[13];
  const float* g1   = (const float*)d_in[14];
  const float* b1   = (const float*)d_in[15];
  const float* Wf1  = (const float*)d_in[16];
  const float* bf1  = (const float*)d_in[17];
  const float* Wf2  = (const float*)d_in[18];
  const float* bf2  = (const float*)d_in[19];
  const float* g2   = (const float*)d_in[20];
  const float* b2   = (const float*)d_in[21];
  const float* hW1  = (const float*)d_in[22];
  const float* hb1  = (const float*)d_in[23];
  const float* hW2  = (const float*)d_in[24];
  const float* hb2  = (const float*)d_in[25];

  char* ws = (char*)d_ws;
  size_t off = 0;
  auto alloc = [&](size_t bytes) -> char* {
    char* p = ws + off;
    off = (off + bytes + 255) & ~(size_t)255;
    return p;
  };
  float*          hf   = (float*)         alloc((size_t)M_ * HID_ * 4);
  float*          proj = (float*)         alloc((size_t)M_ * HID_ * 4);
  float*          r1   = (float*)         alloc((size_t)M_ * 256 * 4);
  unsigned short* hb   = (unsigned short*)alloc((size_t)M_ * HID_ * 2);
  unsigned short* qb   = (unsigned short*)alloc((size_t)M_ * HID_ * 2);
  unsigned short* kb   = (unsigned short*)alloc((size_t)M_ * HID_ * 2);
  unsigned short* vb   = (unsigned short*)alloc((size_t)M_ * HID_ * 2);
  unsigned short* ao   = (unsigned short*)alloc((size_t)M_ * HID_ * 2);
  unsigned short* mid  = (unsigned short*)alloc((size_t)M_ * FF_ * 2);
  unsigned short* kp   = (unsigned short*)alloc((size_t)B_ * NH_ * SP_ * DH_ * 2);
  unsigned short* vt   = (unsigned short*)alloc((size_t)B_ * NH_ * DH_ * SP_ * 2);
  unsigned short* wtq  = (unsigned short*)alloc((size_t)HID_ * HID_ * 2);
  unsigned short* wtk  = (unsigned short*)alloc((size_t)HID_ * HID_ * 2);
  unsigned short* wtv  = (unsigned short*)alloc((size_t)HID_ * HID_ * 2);
  unsigned short* wto  = (unsigned short*)alloc((size_t)HID_ * HID_ * 2);
  unsigned short* wtf1 = (unsigned short*)alloc((size_t)FF_ * HID_ * 2);
  unsigned short* wtf2 = (unsigned short*)alloc((size_t)HID_ * FF_ * 2);
  unsigned short* wth1 = (unsigned short*)alloc((size_t)256 * HID_ * 2);

  const dim3 blk(256);
  auto cdiv = [](int a, int b) { return (a + b - 1) / b; };

  // Embedding + LN
  embed_ln<<<M_, blk, 0, stream>>>(ids, we, pe, eg, eb, hf, hb);

  for (int l = 0; l < L_; ++l) {
    const float* Wq_l = Wq + (size_t)l * HID_ * HID_;
    const float* Wk_l = Wk + (size_t)l * HID_ * HID_;
    const float* Wv_l = Wv + (size_t)l * HID_ * HID_;
    const float* Wo_l = Wo + (size_t)l * HID_ * HID_;
    const float* Wf1_l = Wf1 + (size_t)l * HID_ * FF_;
    const float* Wf2_l = Wf2 + (size_t)l * FF_ * HID_;

    int nqk = HID_ * HID_;
    transpose_conv<<<cdiv(nqk, 256), blk, 0, stream>>>(Wq_l, wtq, HID_, HID_);
    transpose_conv<<<cdiv(nqk, 256), blk, 0, stream>>>(Wk_l, wtk, HID_, HID_);
    transpose_conv<<<cdiv(nqk, 256), blk, 0, stream>>>(Wv_l, wtv, HID_, HID_);
    transpose_conv<<<cdiv(nqk, 256), blk, 0, stream>>>(Wo_l, wto, HID_, HID_);

    dim3 gQ(HID_ / 64, M_ / 128);
    wmma_gemm<0><<<gQ, blk, 0, stream>>>(hb, wtq, bq + l * HID_, qb,
                                         M_, HID_, HID_, 0.125f);  // 1/sqrt(64)
    wmma_gemm<0><<<gQ, blk, 0, stream>>>(hb, wtk, bk + l * HID_, kb,
                                         M_, HID_, HID_, 1.f);
    wmma_gemm<0><<<gQ, blk, 0, stream>>>(hb, wtv, bv + l * HID_, vb,
                                         M_, HID_, HID_, 1.f);

    int npad = B_ * NH_ * SP_ * DH_;
    pad_k<<<cdiv(npad, 256), blk, 0, stream>>>(kb, kp);
    transpose_vp<<<cdiv(npad, 256), blk, 0, stream>>>(vb, vt);

    sw_attn<<<(B_ * NH_ * NB_ * (W_ / 16)) / 8, blk, 0, stream>>>(qb, kp, vt, am, ao);

    wmma_gemm<1><<<gQ, blk, 0, stream>>>(ao, wto, bo + l * HID_, proj,
                                         M_, HID_, HID_, 1.f);
    add_ln<<<M_, blk, 0, stream>>>(hf, proj, g1 + l * HID_, b1 + l * HID_, hb);

    transpose_conv<<<cdiv(HID_ * FF_, 256), blk, 0, stream>>>(Wf1_l, wtf1, HID_, FF_);
    wmma_gemm<2><<<dim3(FF_ / 64, M_ / 128), blk, 0, stream>>>(
        hb, wtf1, bf1 + l * FF_, mid, M_, FF_, HID_, 1.f);

    transpose_conv<<<cdiv(FF_ * HID_, 256), blk, 0, stream>>>(Wf2_l, wtf2, FF_, HID_);
    wmma_gemm<1><<<gQ, blk, 0, stream>>>(mid, wtf2, bf2 + l * HID_, proj,
                                         M_, HID_, FF_, 1.f);
    add_ln<<<M_, blk, 0, stream>>>(hf, proj, g2 + l * HID_, b2 + l * HID_, hb);
  }

  // Classifier head
  transpose_conv<<<cdiv(HID_ * 256, 256), blk, 0, stream>>>(hW1, wth1, HID_, 256);
  wmma_gemm<3><<<dim3(256 / 64, M_ / 128), blk, 0, stream>>>(
      hb, wth1, hb1, r1, M_, 256, HID_, 1.f);
  head2<<<M_, dim3(64), 0, stream>>>(r1, hW2, hb2, (float*)d_out);
}
